// CausalSelfAttention_79680233276057
// MI455X (gfx1250) — matrix operand order repaired and checked by reading, not compiled, not run on hardware
//
#include <hip/hip_runtime.h>

// ---------------------------------------------------------------------------
// CDNA5 (gfx1250) causal self-attention forward.
// B=2, T=2048, C=1024, nh=16, hs=64.  All GEMM-like work on
// v_wmma_f32_16x16x32_bf16 (fp32 accumulate); softmax / RoPE in fp32.
// ---------------------------------------------------------------------------

typedef __attribute__((ext_vector_type(16))) __bf16 v16bf;
typedef __attribute__((ext_vector_type(8)))  float  v8f;

union BFrag { v16bf v; unsigned short u[16]; uint4 q[2]; };

__device__ __forceinline__ unsigned short f2bf(float f) {
  unsigned u = __float_as_uint(f);
  unsigned r = u + 0x7FFFu + ((u >> 16) & 1u);   // round-to-nearest-even
  return (unsigned short)(r >> 16);
}
__device__ __forceinline__ unsigned pk2bf(float lo, float hi) {
  return (unsigned)f2bf(lo) | ((unsigned)f2bf(hi) << 16);
}
__device__ __forceinline__ float bf2f(unsigned short h) {
  return __uint_as_float(((unsigned)h) << 16);
}
__device__ __forceinline__ v8f wmma_bf16(const v16bf a, const v16bf b, const v8f c) {
  return __builtin_amdgcn_wmma_f32_16x16x32_bf16(false, a, false, b, (short)0, c,
                                                 false, false);
}

#define N_EMBED 1024
#define N_HEAD  16
#define HS      64
#define TSEQ    2048
#define BATCH   2
#define MTOT    (BATCH * TSEQ)          // 4096
#define PITCH   40                      // LDS row pitch (ushorts) = 80B, 16B-aligned

// ---------------------------------------------------------------------------
// Generic bf16 WMMA GEMM core: block 256 thr (8 waves), tile 64(M) x 256(N),
// K-step 32, wave tile 32x64 (2x4 WMMA tiles => 8 wmma / k-step / wave).
// A staged [m][k] (vector b128), B staged transposed [n][k] (b64 stores so
// fragments are contiguous b128 loads).
// ---------------------------------------------------------------------------

// QKV GEMM: x[4096x1024] @ w_attn[1024x3072] + bias; scatter epilogue:
//   q -> [bh][T][hs], k -> [bh][T][hs], v -> transposed [bh][hs][T]  (bf16)
__global__ __launch_bounds__(256) void qkv_gemm_kernel(
    const float* __restrict__ x, const float* __restrict__ w,
    const float* __restrict__ bias, unsigned short* __restrict__ qb,
    unsigned short* __restrict__ kb, unsigned short* __restrict__ vT) {
  __shared__ unsigned short lA[64 * PITCH];    //  5.1 KB
  __shared__ unsigned short lB[256 * PITCH];   // 20.5 KB

  const int tid  = threadIdx.x;
  const int wave = tid >> 5, lane = tid & 31;
  const int rlo = lane & 15, half = lane >> 4;
  const int mBase = blockIdx.x * 64;
  const int nBase = blockIdx.y * 256;
  const int wm = (wave >> 2) * 32;
  const int wn = (wave & 3) * 64;

  // per-thread staging coords: A: row = tid>>2, 8 k's at (tid&3)*8 ; B: n = tid
  const int am = tid >> 2, ak = (tid & 3) * 8;

  v8f acc[2][4] = {};

  for (int k0 = 0; k0 < N_EMBED; k0 += 32) {
    // ---- batched global loads into registers ----
    const float* ax = &x[(size_t)(mBase + am) * N_EMBED + k0 + ak];
    float4 a0 = *(const float4*)ax;
    float4 a1 = *(const float4*)(ax + 4);
    float breg[32];
#pragma unroll
    for (int kk = 0; kk < 32; ++kk)
      breg[kk] = w[(size_t)(k0 + kk) * (3 * N_EMBED) + nBase + tid];

    // ---- convert + vector LDS stores ----
    uint4 ap;
    ap.x = pk2bf(a0.x, a0.y); ap.y = pk2bf(a0.z, a0.w);
    ap.z = pk2bf(a1.x, a1.y); ap.w = pk2bf(a1.z, a1.w);
    *(uint4*)&lA[am * PITCH + ak] = ap;
#pragma unroll
    for (int g = 0; g < 8; ++g) {
      uint2 bp;
      bp.x = pk2bf(breg[g * 4 + 0], breg[g * 4 + 1]);
      bp.y = pk2bf(breg[g * 4 + 2], breg[g * 4 + 3]);
      *(uint2*)&lB[tid * PITCH + g * 4] = bp;   // transposed [n][k]
    }
    if (k0 + 32 < N_EMBED)
      __builtin_prefetch(ax + 32, 0, 1);        // global_prefetch_b8
    __syncthreads();

    // ---- fragments: all contiguous b128 LDS loads ----
    BFrag a[2], b[4];
#pragma unroll
    for (int r = 0; r < 2; ++r) {
      const unsigned short* s = &lA[(wm + r * 16 + rlo) * PITCH + half * 8];
      a[r].q[0] = *(const uint4*)s;
      a[r].q[1] = *(const uint4*)(s + 16);
    }
#pragma unroll
    for (int c = 0; c < 4; ++c) {
      const unsigned short* s = &lB[(wn + c * 16 + rlo) * PITCH + half * 8];
      b[c].q[0] = *(const uint4*)s;
      b[c].q[1] = *(const uint4*)(s + 16);
    }
#pragma unroll
    for (int r = 0; r < 2; ++r)
#pragma unroll
      for (int c = 0; c < 4; ++c)
        acc[r][c] = wmma_bf16(a[r].v, b[c].v, acc[r][c]);
    __syncthreads();
  }

  // ---- epilogue: + bias, scatter bf16 to q / k / v^T ----
#pragma unroll
  for (int r = 0; r < 2; ++r) {
#pragma unroll
    for (int c = 0; c < 4; ++c) {
      int n = nBase + wn + c * 16 + rlo;
      float bv = bias[n];
      int region = n >> 10;            // 0:q 1:k 2:v
      int h = (n & 1023) >> 6;
      int d = n & 63;
#pragma unroll
      for (int e = 0; e < 8; ++e) {
        int m = mBase + wm + r * 16 + e + half * 8;
        int bidx = m >> 11, t = m & 2047;
        unsigned short hv = f2bf(acc[r][c][e] + bv);
        size_t bh = (size_t)(bidx * N_HEAD + h);
        if (region == 0)      qb[(bh * TSEQ + t) * HS + d] = hv;
        else if (region == 1) kb[(bh * TSEQ + t) * HS + d] = hv;
        else                  vT[(bh * HS + d) * TSEQ + t] = hv;
      }
    }
  }
}

// ---------------------------------------------------------------------------
// In-place RoPE on q and k (bf16): one thread per (buffer, bh, t, pair j).
// Per-buffer pair count = 32*2048*32 = 2^21.  sincos computed directly.
// ---------------------------------------------------------------------------
__global__ __launch_bounds__(256) void rope_apply_kernel(
    unsigned int* __restrict__ qb, unsigned int* __restrict__ kb) {
  int idx = blockIdx.x * 256 + threadIdx.x;          // [0, 2^22)
  unsigned int* p = (idx >> 21) ? kb : qb;
  int r = idx & 0x1FFFFF;
  int j = r & 31;                                    // pair index (d = 2j)
  int t = (r >> 5) & 2047;
  float theta = __powf(10000.0f, -(float)j * (1.0f / 512.0f));
  float a = (float)t * theta;
  float sn, cs;
  __sincosf(a, &sn, &cs);
  unsigned v = p[r];
  float xr = bf2f((unsigned short)(v & 0xFFFF));
  float xi = bf2f((unsigned short)(v >> 16));
  p[r] = pk2bf(xr * cs - xi * sn, xi * cs + xr * sn);
}

// ---------------------------------------------------------------------------
// Flash attention per (b,h): block = 128 thr (4 independent waves), each wave
// owns 16 query rows, streams 32 keys/iter.  Q/K pre-rotated.  Online softmax
// with shfl_xor row reductions (rows live in 16-lane halves of the C layout);
// P routed through per-wave LDS (cross-lane write->read ordered by
// s_wait_dscnt); PV from transposed V gives contiguous b128 fragments.
// No block barriers: waves have divergent causal trip counts.
// ---------------------------------------------------------------------------
__global__ __launch_bounds__(128) void attn_kernel(
    const unsigned short* __restrict__ qb, const unsigned short* __restrict__ kb,
    const unsigned short* __restrict__ vT, unsigned short* __restrict__ yb) {
  const int bh   = blockIdx.x;                       // 0..31
  const int wave = threadIdx.x >> 5, lane = threadIdx.x & 31;
  const int rlo = lane & 15, half = lane >> 4;
  const int q0 = blockIdx.y * 64 + wave * 16;
  const unsigned short* Q = qb + (size_t)bh * TSEQ * HS;
  const unsigned short* K = kb + (size_t)bh * TSEQ * HS;
  const unsigned short* V = vT + (size_t)bh * HS * TSEQ;

  __shared__ unsigned short ldsP[4][16 * PITCH];     // 5 KB, per-wave slices

  // Q A-fragments (16x32 bf16, two d-halves), pure vector loads
  const int tq = q0 + rlo;
  BFrag aQ[2];
#pragma unroll
  for (int f = 0; f < 2; ++f) {
    const unsigned short* s = &Q[(size_t)tq * HS + f * 32 + half * 8];
    aQ[f].q[0] = *(const uint4*)s;
    aQ[f].q[1] = *(const uint4*)(s + 16);
  }

  float mrow[8], lrow[8];
#pragma unroll
  for (int v = 0; v < 8; ++v) { mrow[v] = -3.0e38f; lrow[v] = 0.0f; }
  v8f o[4] = {};

  for (int kt = 0; kt <= q0 + 15; kt += 32) {
    // K B-fragments (2 key halves x 2 d halves): contiguous b128 loads
    BFrag bK[2][2];
#pragma unroll
    for (int h2 = 0; h2 < 2; ++h2) {
      int tk = kt + h2 * 16 + rlo;
#pragma unroll
      for (int f = 0; f < 2; ++f) {
        const unsigned short* s = &K[(size_t)tk * HS + f * 32 + half * 8];
        bK[h2][f].q[0] = *(const uint4*)s;
        bK[h2][f].q[1] = *(const uint4*)(s + 16);
      }
    }
    // V B-fragments for all four output d-tiles
    BFrag bV[4];
#pragma unroll
    for (int ot = 0; ot < 4; ++ot) {
      const unsigned short* s = &V[(size_t)(ot * 16 + rlo) * TSEQ + kt + half * 8];
      bV[ot].q[0] = *(const uint4*)s;
      bV[ot].q[1] = *(const uint4*)(s + 16);
    }

    // S = Q K^T : two 16x16 key tiles, K=64 via two WMMAs each
    v8f s0 = {}, s1 = {};
    s0 = wmma_bf16(aQ[0].v, bK[0][0].v, s0);
    s0 = wmma_bf16(aQ[1].v, bK[0][1].v, s0);
    s1 = wmma_bf16(aQ[0].v, bK[1][0].v, s1);
    s1 = wmma_bf16(aQ[1].v, bK[1][1].v, s1);

    // causal mask + scale + online softmax
    const float scale = 0.125f;                      // 1/sqrt(64)
    float p0[8], p1[8];
#pragma unroll
    for (int v = 0; v < 8; ++v) {
      int trow = q0 + v + half * 8;
      float e0 = (kt + rlo      <= trow) ? s0[v] * scale : -3.0e38f;
      float e1 = (kt + 16 + rlo <= trow) ? s1[v] * scale : -3.0e38f;
      float mx = fmaxf(e0, e1);
#pragma unroll
      for (int off = 8; off; off >>= 1) mx = fmaxf(mx, __shfl_xor(mx, off, 32));
      float mnew  = fmaxf(mrow[v], mx);
      float alpha = __expf(mrow[v] - mnew);
      e0 = __expf(e0 - mnew);
      e1 = __expf(e1 - mnew);
      float sum = e0 + e1;
#pragma unroll
      for (int off = 8; off; off >>= 1) sum += __shfl_xor(sum, off, 32);
      lrow[v] = lrow[v] * alpha + sum;
      mrow[v] = mnew;
      p0[v] = e0; p1[v] = e1;
#pragma unroll
      for (int ot = 0; ot < 4; ++ot) o[ot][v] *= alpha;
    }

    // P: C-layout -> A-layout via per-wave LDS slice
    unsigned short* Pl = ldsP[wave];
#pragma unroll
    for (int v = 0; v < 8; ++v) {
      int M = v + half * 8;
      Pl[M * PITCH + rlo]      = f2bf(p0[v]);
      Pl[M * PITCH + 16 + rlo] = f2bf(p1[v]);
    }
    asm volatile("s_wait_dscnt 0" ::: "memory");     // cross-lane LDS ordering
    BFrag aP;
    {
      const unsigned short* s = &Pl[rlo * PITCH + half * 8];
      aP.q[0] = *(const uint4*)s;
      aP.q[1] = *(const uint4*)(s + 16);
    }

    // O += P V
#pragma unroll
    for (int ot = 0; ot < 4; ++ot)
      o[ot] = wmma_bf16(aP.v, bV[ot].v, o[ot]);
  }

  // normalize + write y (bf16) in [B][T][C] layout
  const int b = bh >> 4, h = bh & 15;
#pragma unroll
  for (int ot = 0; ot < 4; ++ot) {
#pragma unroll
    for (int v = 0; v < 8; ++v) {
      int trow = q0 + v + half * 8;
      float val = o[ot][v] / lrow[v];
      yb[(size_t)(b * TSEQ + trow) * N_EMBED + h * HS + ot * 16 + rlo] = f2bf(val);
    }
  }
}

// ---------------------------------------------------------------------------
// Output projection: bf16 y [4096x1024] @ bf16(w_proj[1024x1024]) + bias
// -> fp32 out.  Same tiling as QKV (block 64x256, wave 32x64).
// ---------------------------------------------------------------------------
__global__ __launch_bounds__(256) void proj_gemm_kernel(
    const unsigned short* __restrict__ yb, const float* __restrict__ w,
    const float* __restrict__ bias, float* __restrict__ out) {
  __shared__ unsigned short lA[64 * PITCH];
  __shared__ unsigned short lB[256 * PITCH];

  const int tid  = threadIdx.x;
  const int wave = tid >> 5, lane = tid & 31;
  const int rlo = lane & 15, half = lane >> 4;
  const int mBase = blockIdx.x * 64;
  const int nBase = blockIdx.y * 256;
  const int wm = (wave >> 2) * 32;
  const int wn = (wave & 3) * 64;
  const int am = tid >> 2, ak = (tid & 3) * 8;

  v8f acc[2][4] = {};

  for (int k0 = 0; k0 < N_EMBED; k0 += 32) {
    const unsigned short* ax = &yb[(size_t)(mBase + am) * N_EMBED + k0 + ak];
    uint4 ap = *(const uint4*)ax;                    // already bf16
    float breg[32];
#pragma unroll
    for (int kk = 0; kk < 32; ++kk)
      breg[kk] = w[(size_t)(k0 + kk) * N_EMBED + nBase + tid];

    *(uint4*)&lA[am * PITCH + ak] = ap;
#pragma unroll
    for (int g = 0; g < 8; ++g) {
      uint2 bp;
      bp.x = pk2bf(breg[g * 4 + 0], breg[g * 4 + 1]);
      bp.y = pk2bf(breg[g * 4 + 2], breg[g * 4 + 3]);
      *(uint2*)&lB[tid * PITCH + g * 4] = bp;
    }
    if (k0 + 32 < N_EMBED)
      __builtin_prefetch(ax + 32, 0, 1);
    __syncthreads();

    BFrag a[2], b[4];
#pragma unroll
    for (int r = 0; r < 2; ++r) {
      const unsigned short* s = &lA[(wm + r * 16 + rlo) * PITCH + half * 8];
      a[r].q[0] = *(const uint4*)s;
      a[r].q[1] = *(const uint4*)(s + 16);
    }
#pragma unroll
    for (int c = 0; c < 4; ++c) {
      const unsigned short* s = &lB[(wn + c * 16 + rlo) * PITCH + half * 8];
      b[c].q[0] = *(const uint4*)s;
      b[c].q[1] = *(const uint4*)(s + 16);
    }
#pragma unroll
    for (int r = 0; r < 2; ++r)
#pragma unroll
      for (int c = 0; c < 4; ++c)
        acc[r][c] = wmma_bf16(a[r].v, b[c].v, acc[r][c]);
    __syncthreads();
  }

#pragma unroll
  for (int r = 0; r < 2; ++r) {
#pragma unroll
    for (int c = 0; c < 4; ++c) {
      int n = nBase + wn + c * 16 + rlo;
      float bv = bias[n];
#pragma unroll
      for (int e = 0; e < 8; ++e) {
        int m = mBase + wm + r * 16 + e + half * 8;
        out[(size_t)m * N_EMBED + n] = acc[r][c][e] + bv;
      }
    }
  }
}

// ---------------------------------------------------------------------------
// Launcher.  Workspace (32 MB):
//   qb @ 0     : 8 MB bf16 [32][2048][64]
//   kb @ 8 MB  : 8 MB bf16 [32][2048][64]
//   vT @ 16 MB : 8 MB bf16 [32][64][2048]
//   yb @ 24 MB : 8 MB bf16 [2][2048][1024]
// ---------------------------------------------------------------------------
extern "C" void kernel_launch(void* const* d_in, const int* in_sizes, int n_in,
                              void* d_out, int out_size, void* d_ws,
                              size_t ws_size, hipStream_t stream) {
  const float* x      = (const float*)d_in[0];
  const float* w_attn = (const float*)d_in[1];
  const float* b_attn = (const float*)d_in[2];
  const float* w_proj = (const float*)d_in[3];
  const float* b_proj = (const float*)d_in[4];
  float* out = (float*)d_out;

  char* ws = (char*)d_ws;
  const size_t SEG = (size_t)8 << 20;
  unsigned short* qb = (unsigned short*)(ws);
  unsigned short* kb = (unsigned short*)(ws + SEG);
  unsigned short* vT = (unsigned short*)(ws + 2 * SEG);
  unsigned short* yb = (unsigned short*)(ws + 3 * SEG);

  qkv_gemm_kernel<<<dim3(MTOT / 64, (3 * N_EMBED) / 256), 256, 0, stream>>>(
      x, w_attn, b_attn, qb, kb, vT);

  rope_apply_kernel<<<(2 * 2097152) / 256, 256, 0, stream>>>(
      (unsigned int*)qb, (unsigned int*)kb);

  attn_kernel<<<dim3(BATCH * N_HEAD, TSEQ / 64), 128, 0, stream>>>(
      qb, kb, vT, yb);

  proj_gemm_kernel<<<dim3(MTOT / 64, N_EMBED / 256), 256, 0, stream>>>(
      yb, w_proj, b_proj, out);
}
